// LSTMPredictor_84198538871066
// MI455X (gfx1250) — matrix operand order repaired and checked by reading, not compiled, not run on hardware
//
#include <hip/hip_runtime.h>
#include <hip/hip_bf16.h>

// LSTM predictor for MI455X (gfx1250), wave32 + WMMA f16->f32.
// H=128, L=5, B=32, T=512, future=18, OUT=16  =>  3 autoregressive iterations.

#define H      128
#define B4H    512      // 4*H
#define NB     32       // batch
#define T0     512
#define TMAX   514
#define NLAY   5
#define OUTSZ  16

typedef __attribute__((ext_vector_type(16))) _Float16 v16h;
typedef __attribute__((ext_vector_type(8)))  _Float16 v8h;
typedef __attribute__((ext_vector_type(8)))  float    v8f;

#define H_PAD   136   // padded LDS row stride (f16) for h (128 + 8)

__device__ __forceinline__ float sigm_(float x) { return 1.0f / (1.0f + __expf(-x)); }
__device__ __forceinline__ float tanh_(float x) { return 2.0f / (1.0f + __expf(-2.0f * x)) - 1.0f; }

__device__ __forceinline__ v16h mkfrag(v8h lo, v8h hi) {
    v16h r;
#pragma unroll
    for (int i = 0; i < 8; ++i) { r[i] = lo[i]; r[i + 8] = hi[i]; }
    return r;
}

// ---------------------------------------------------------------------------
// Prep: f32->f16 weight conversion + bias sum (b_ih + b_hh)
// ---------------------------------------------------------------------------
__global__ void prep_kernel(const float* __restrict__ whh_f,   // [5][512][128]
                            const float* __restrict__ wih_f,   // [4][512][128]
                            const float* __restrict__ bih,     // [5][512]
                            const float* __restrict__ bhh,     // [5][512]
                            _Float16* __restrict__ whh_h,
                            _Float16* __restrict__ wih_h,
                            float* __restrict__ bsum) {
    const int N1 = NLAY * B4H * H;          // 327680
    const int N2 = (NLAY - 1) * B4H * H;    // 262144
    const int N3 = NLAY * B4H;              // 2560
    int idx = blockIdx.x * blockDim.x + threadIdx.x;
    int total = N1 + N2 + N3;
    for (; idx < total; idx += gridDim.x * blockDim.x) {
        if (idx < N1) {
            whh_h[idx] = (_Float16)whh_f[idx];
        } else if (idx < N1 + N2) {
            int i = idx - N1;
            wih_h[i] = (_Float16)wih_f[i];
        } else {
            int i = idx - N1 - N2;
            bsum[i] = bih[i] + bhh[i];
        }
    }
}

// Copy x (B,512) into extended buffer x_ext (B,514)
__global__ void xinit_kernel(const float* __restrict__ x, float* __restrict__ x_ext) {
    int idx = blockIdx.x * blockDim.x + threadIdx.x;
    if (idx < NB * T0) {
        int b = idx / T0, t = idx - b * T0;
        x_ext[b * TMAX + t] = x[idx];
    }
}

// ---------------------------------------------------------------------------
// xp is stored in WMMA C-fragment tiled layout:
//   xp[((t*64 + nt*2 + mt) * 32 + lane) * 8 + r]
// where nt = gate-col tile (0..31), mt = batch tile (0..1),
//       lane = n + 16*h1, element r -> row m = r + 8*h1 (+16*mt).
// ---------------------------------------------------------------------------

// Layer-0 input projection: value = x_ext[b][t]*W_ih0[g] + bsum0[g]
__global__ void xp_l0_kernel(const float* __restrict__ x_ext,
                             const float* __restrict__ wih0,   // [512]
                             const float* __restrict__ bsum0,  // [512]
                             float* __restrict__ xp, int T) {
    int idx = blockIdx.x * blockDim.x + threadIdx.x;
    int total = T * 64 * 32 * 8;  // = T*32*512
    for (; idx < total; idx += gridDim.x * blockDim.x) {
        int r    = idx & 7;
        int lane = (idx >> 3) & 31;
        int tile = idx >> 8;
        int t    = tile >> 6;
        int rem  = tile & 63;
        int nt   = rem >> 1;
        int mt   = rem & 1;
        int h1   = lane >> 4;
        int b    = mt * 16 + r + 8 * h1;
        int g    = nt * 16 + (lane & 15);
        xp[idx] = x_ext[b * TMAX + t] * wih0[g] + bsum0[g];
    }
}

// ---------------------------------------------------------------------------
// xp GEMM (layers 1..4): xp = seq[M][128](f16) @ W_ih^T + bias, M = 32*T.
// One block = 256 threads = 8 waves; each wave: 1 mtile x 4 ntiles.
// blockIdx.x = t*2 + mt.
// ---------------------------------------------------------------------------
__global__ __launch_bounds__(256) void xp_gemm_kernel(
        const _Float16* __restrict__ a_seq,  // [M][128]
        const _Float16* __restrict__ wih,    // [512][128] (row g, col k)
        const float*    __restrict__ bsum,   // [512]
        float*          __restrict__ xp) {   // tiled layout, see above
    const int lane  = threadIdx.x & 31;
    const int wave  = threadIdx.x >> 5;      // 0..7 -> ntile group
    const int h1    = lane >> 4;
    const int l15   = lane & 15;
    const int mtile = blockIdx.x;

    __builtin_prefetch(a_seq + (size_t)(mtile * 16 + l15) * H, 0, 3);

    v8f acc[4];
#pragma unroll
    for (int j = 0; j < 4; ++j) {
        float bv = bsum[(wave * 4 + j) * 16 + l15];
#pragma unroll
        for (int r = 0; r < 8; ++r) acc[j][r] = bv;
    }

#pragma unroll
    for (int kt = 0; kt < 4; ++kt) {
        const _Float16* arow = a_seq + (size_t)(mtile * 16 + l15) * H + kt * 32 + 8 * h1;
        v16h a = mkfrag(*(const v8h*)arow, *(const v8h*)(arow + 16));
#pragma unroll
        for (int j = 0; j < 4; ++j) {
            int nt = wave * 4 + j;
            const _Float16* brow = wih + (size_t)(nt * 16 + l15) * H + kt * 32 + 8 * h1;
            v16h b = mkfrag(*(const v8h*)brow, *(const v8h*)(brow + 16));
            acc[j] = __builtin_amdgcn_wmma_f32_16x16x32_f16(
                false, a, false, b, (short)0, acc[j], false, false);
        }
    }

    const int t  = mtile >> 1;
    const int mt = mtile & 1;
#pragma unroll
    for (int j = 0; j < 4; ++j) {
        int nt = wave * 4 + j;
        size_t tile = (size_t)t * 64 + nt * 2 + mt;
        *(v8f*)(xp + (tile * 32 + lane) * 8) = acc[j];
    }
}

// ---------------------------------------------------------------------------
// Recurrent scan: one workgroup, 512 threads = 16 waves.
// Wave w: mtile = w>>3 (batch rows), jt = w&7 (hidden col tile). It computes
// the (mtile,jt) tile of all four gates, so the c/h update is wave-local.
// W_hh B-fragments are REGISTER-resident (16 x v16h = 128 VGPRs per wave).
// h is double-buffered in LDS -> only ONE barrier per timestep:
//   step t reads h(t-1) from buf[(t+1)&1], writes h(t) to buf[t&1].
// ---------------------------------------------------------------------------
__global__ __launch_bounds__(512) void lstm_scan_kernel(
        const float*    __restrict__ xp,      // tiled layout
        const _Float16* __restrict__ whh,     // [512][128] (this layer, f16)
        _Float16*       __restrict__ out_seq, // [T*32][128]
        float*          __restrict__ hlast,   // [32][128]
        int T) {
    __shared__ _Float16 hlds[2][NB * H_PAD];

    const int tid   = threadIdx.x;
    const int lane  = tid & 31;
    const int wave  = tid >> 5;
    const int h1    = lane >> 4;
    const int l15   = lane & 15;
    const int mtile = wave >> 3;   // 0..1
    const int jt    = wave & 7;    // hidden tile 0..7

    // Load this wave's W_hh B-fragments into registers (loop-invariant).
    v16h bfr[16];
#pragma unroll
    for (int g = 0; g < 4; ++g) {
#pragma unroll
        for (int kt = 0; kt < 4; ++kt) {
            const _Float16* brow =
                whh + (size_t)((g * 8 + jt) * 16 + l15) * H + kt * 32 + 8 * h1;
            bfr[g * 4 + kt] = mkfrag(*(const v8h*)brow, *(const v8h*)(brow + 16));
        }
    }

    // buf[1] holds h(-1) = 0 (step 0 reads buf[(0+1)&1] = buf[1])
    for (int i = tid; i < NB * H_PAD; i += 512) hlds[1][i] = (_Float16)0.0f;
    __syncthreads();

    v8f c;
#pragma unroll
    for (int r = 0; r < 8; ++r) c[r] = 0.0f;

    // Preload xp gate tiles for t=0
    v8f xpv[4];
#pragma unroll
    for (int g = 0; g < 4; ++g)
        xpv[g] = *(const v8f*)(xp +
            ((size_t)((g * 8 + jt) * 2 + mtile) * 32 + lane) * 8);

    for (int t = 0; t < T; ++t) {
        const _Float16* hin  = hlds[(t + 1) & 1];
        _Float16*       hout = hlds[t & 1];

        // A fragments (h_prev) from LDS: 4 ktiles covering K=128
        v16h a[4];
#pragma unroll
        for (int kt = 0; kt < 4; ++kt) {
            const _Float16* row = hin + (mtile * 16 + l15) * H_PAD + kt * 32 + 8 * h1;
            a[kt] = mkfrag(*(const v8h*)row, *(const v8h*)(row + 16));
        }

        v8f acc[4];  // gates i,f,g,o tile (mtile, jt)
#pragma unroll
        for (int g = 0; g < 4; ++g) {
            acc[g] = xpv[g];
#pragma unroll
            for (int kt = 0; kt < 4; ++kt) {
                acc[g] = __builtin_amdgcn_wmma_f32_16x16x32_f16(
                    false, a[kt], false, bfr[g * 4 + kt], (short)0, acc[g], false, false);
            }
        }

        // Prefetch next timestep's xp tiles (overlaps elementwise + barrier)
        {
            int tn = (t + 1 < T) ? (t + 1) : t;
#pragma unroll
            for (int g = 0; g < 4; ++g)
                xpv[g] = *(const v8f*)(xp +
                    ((size_t)(tn * 64 + (g * 8 + jt) * 2 + mtile) * 32 + lane) * 8);
        }

        // Elementwise LSTM cell update (wave-local tile); writes go to the
        // OTHER buffer than the one being read this step -> no barrier needed
        // between the A-fragment reads above and these writes.
#pragma unroll
        for (int r = 0; r < 8; ++r) {
            float iS = sigm_(acc[0][r]);
            float fS = sigm_(acc[1][r]);
            float gT = tanh_(acc[2][r]);
            float oS = sigm_(acc[3][r]);
            float cn = fS * c[r] + iS * gT;
            c[r] = cn;
            float hn = oS * tanh_(cn);
            int m   = mtile * 16 + r + 8 * h1;
            int col = jt * 16 + l15;
            hout[m * H_PAD + col] = (_Float16)hn;
            if (t == T - 1) hlast[m * H + col] = hn;
        }

        __syncthreads();  // h(t) visible to all before step t+1 reads it

        // Cooperative vector store of h(t) to out_seq (16B per thread).
        // Reads buf[t&1]; step t+1 writes buf[(t+1)&1] -> no conflict.
        {
            int m  = tid >> 4;   // 0..31
            int ch = tid & 15;   // 16B chunk
            v8h v = *(const v8h*)(hout + m * H_PAD + ch * 8);
            *(v8h*)(out_seq + (size_t)(t * NB + m) * H + ch * 8) = v;
        }
    }
}

// ---------------------------------------------------------------------------
// FC head + output assembly (fused): z=relu(h_last); z=relu(z@W1+b1);
// fo=z@W2+b2; then scatter into d_out / extend x. Tiny, 1 block.
// ---------------------------------------------------------------------------
__global__ __launch_bounds__(128) void fc_head_kernel(
        const float* __restrict__ hlast, const float* __restrict__ W1,
        const float* __restrict__ b1,    const float* __restrict__ W2,
        const float* __restrict__ b2,    float* __restrict__ out,
        float* __restrict__ x_ext, int iter, int Tcur) {
    __shared__ float z1[NB][H];
    __shared__ float z2[NB][H];
    __shared__ float fo[NB * OUTSZ];
    int j = threadIdx.x;
    for (int b = 0; b < NB; ++b) z1[b][j] = fmaxf(hlast[b * H + j], 0.0f);
    __syncthreads();
    for (int b = 0; b < NB; ++b) {
        float s = b1[j];
#pragma unroll 4
        for (int k = 0; k < H; ++k) s += z1[b][k] * W1[k * H + j];
        z2[b][j] = fmaxf(s, 0.0f);
    }
    __syncthreads();
    for (int idx = j; idx < NB * OUTSZ; idx += 128) {
        int b = idx >> 4, col = idx & 15;
        float s = b2[col];
#pragma unroll 4
        for (int k = 0; k < H; ++k) s += z2[b][k] * W2[k * OUTSZ + col];
        fo[idx] = s;
    }
    __syncthreads();
    if (iter < 2) {
        if (j < NB) {
            float v = fo[j * OUTSZ + (OUTSZ - 1)];
            out[j * 18 + iter] = v;
            x_ext[j * TMAX + Tcur] = v;
        }
    } else {
        for (int idx = j; idx < NB * OUTSZ; idx += 128) {
            int b = idx >> 4, col = idx & 15;
            out[b * 18 + 2 + col] = fo[idx];
        }
    }
}

// ---------------------------------------------------------------------------
extern "C" void kernel_launch(void* const* d_in, const int* in_sizes, int n_in,
                              void* d_out, int out_size, void* d_ws, size_t ws_size,
                              hipStream_t stream) {
    (void)in_sizes; (void)n_in; (void)out_size; (void)ws_size;
    const float* x      = (const float*)d_in[0];
    // d_in[1] = future (=18, fixed by problem sizes)
    const float* W_ih0  = (const float*)d_in[2];   // (512,1)
    const float* W_ihR  = (const float*)d_in[3];   // (4,512,128)
    const float* W_hh   = (const float*)d_in[4];   // (5,512,128)
    const float* b_ih   = (const float*)d_in[5];   // (5,512)
    const float* b_hh   = (const float*)d_in[6];   // (5,512)
    const float* W1     = (const float*)d_in[7];
    const float* b1     = (const float*)d_in[8];
    const float* W2     = (const float*)d_in[9];
    const float* b2     = (const float*)d_in[10];
    float* out = (float*)d_out;

    // Workspace carve (256B aligned)
    char* p = (char*)d_ws;
    auto carve = [&p](size_t bytes) -> char* {
        char* r = p; p += (bytes + 255) & ~(size_t)255; return r;
    };
    _Float16* whh_h = (_Float16*)carve(NLAY * B4H * H * sizeof(_Float16));
    _Float16* wih_h = (_Float16*)carve((NLAY - 1) * B4H * H * sizeof(_Float16));
    float*    bsum  = (float*)   carve(NLAY * B4H * sizeof(float));
    float*    x_ext = (float*)   carve(NB * TMAX * sizeof(float));
    float*    xp    = (float*)   carve((size_t)TMAX * NB * B4H * sizeof(float));
    _Float16* seqA  = (_Float16*)carve((size_t)TMAX * NB * H * sizeof(_Float16));
    _Float16* seqB  = (_Float16*)carve((size_t)TMAX * NB * H * sizeof(_Float16));
    float*    hlast = (float*)   carve(NB * H * sizeof(float));

    prep_kernel<<<1024, 256, 0, stream>>>(W_hh, W_ihR, b_ih, b_hh, whh_h, wih_h, bsum);
    xinit_kernel<<<(NB * T0 + 255) / 256, 256, 0, stream>>>(x, x_ext);

    for (int iter = 0; iter < 3; ++iter) {
        const int T = T0 + iter;
        // ---- layer 0 ----
        xp_l0_kernel<<<2048, 256, 0, stream>>>(x_ext, W_ih0, bsum, xp, T);
        lstm_scan_kernel<<<1, 512, 0, stream>>>(xp, whh_h, seqA, hlast, T);
        _Float16* cur = seqA; _Float16* nxt = seqB;
        // ---- layers 1..4 ----
        for (int l = 1; l < NLAY; ++l) {
            xp_gemm_kernel<<<2 * T, 256, 0, stream>>>(
                cur, wih_h + (size_t)(l - 1) * B4H * H, bsum + l * B4H, xp);
            lstm_scan_kernel<<<1, 512, 0, stream>>>(
                xp, whh_h + (size_t)l * B4H * H, nxt, hlast, T);
            _Float16* tmp = cur; cur = nxt; nxt = tmp;
        }
        fc_head_kernel<<<1, 128, 0, stream>>>(hlast, W1, b1, W2, b2, out, x_ext, iter, T);
    }
}
